// BMMS8TS8NS8T_61297773248990
// MI455X (gfx1250) — 8-bit matrix operand order repaired and checked by reading, compiled, not run on hardware
//
#include <hip/hip_runtime.h>

typedef __attribute__((ext_vector_type(8))) int v8i;
typedef __attribute__((ext_vector_type(4))) int v4i;
typedef unsigned long long u64;

#define B_  8
#define H_  12
#define S_  1024
#define D_  64
#define BH_ (B_ * H_)

__device__ __forceinline__ int pack4_i8(int a, int b, int c, int d) {
  return (a & 0xFF) | ((b & 0xFF) << 8) | ((c & 0xFF) << 16) | ((d & 0xFF) << 24);
}

__global__ void pack_x_kernel(const int* __restrict__ x,
                              unsigned char* __restrict__ x8, int n4) {
  int i = blockIdx.x * blockDim.x + threadIdx.x;
  if (i >= n4) return;
  v4i v = ((const v4i*)x)[i];
  ((int*)x8)[i] = pack4_i8(v[0], v[1], v[2], v[3]);
}

__global__ void pack_yT_kernel(const int* __restrict__ y,
                               unsigned char* __restrict__ yT8, int n) {
  int i = blockIdx.x * blockDim.x + threadIdx.x;
  if (i >= n) return;
  int bh  = i / (4 * S_);
  int rem = i % (4 * S_);
  int dq  = rem / S_;
  int t   = rem % S_;
  int d0  = dq * 16;
  const int* base = y + (size_t)bh * D_ * S_ + t;
  int w[4];
#pragma unroll
  for (int q = 0; q < 4; ++q) {
    int a0 = base[(size_t)(d0 + 4 * q + 0) * S_];
    int a1 = base[(size_t)(d0 + 4 * q + 1) * S_];
    int a2 = base[(size_t)(d0 + 4 * q + 2) * S_];
    int a3 = base[(size_t)(d0 + 4 * q + 3) * S_];
    w[q] = pack4_i8(a0, a1, a2, a3);
  }
  v4i out = {w[0], w[1], w[2], w[3]};
  *((v4i*)(yT8 + ((size_t)bh * S_ + t) * D_ + d0)) = out;
}

__global__ void rowsum_kernel(const int* __restrict__ x,
                              int* __restrict__ rsum, int nrows) {
  int r = blockIdx.x * blockDim.x + threadIdx.x;
  if (r >= nrows) return;
  const v4i* p = (const v4i*)(x + (size_t)r * D_);
  int s = 0;
#pragma unroll
  for (int q = 0; q < 16; ++q) {
    v4i v = p[q];
    s += v[0] + v[1] + v[2] + v[3];
  }
  rsum[r] = s;
}

__global__ void colsum_kernel(const int* __restrict__ y,
                              int* __restrict__ csum, int ncols) {
  int i = blockIdx.x * blockDim.x + threadIdx.x;
  if (i >= ncols) return;
  int bh = i / S_;
  int t  = i % S_;
  const int* base = y + (size_t)bh * D_ * S_ + t;
  int s = 0;
#pragma unroll
  for (int d = 0; d < D_; ++d) s += base[(size_t)d * S_];
  csum[i] = s;
}

__global__ void __launch_bounds__(256)
bmm_i8_wmma_kernel(const unsigned char* __restrict__ x8,
                   const unsigned char* __restrict__ yT8,
                   const int* __restrict__ rsum,
                   const int* __restrict__ csum,
                   float* __restrict__ out,
                   const float* __restrict__ alpha_p,
                   const float* __restrict__ a_zp_p,
                   const float* __restrict__ b_zp_p,
                   const float* __restrict__ out_zp_p,
                   const float* __restrict__ o_alpha_p) {
  const int lane = threadIdx.x & 31;
  const int wave = threadIdx.x >> 5;
  const int hf   = lane >> 4;
  const int l15  = lane & 15;

  const int bh   = blockIdx.z;
  const int wRow = blockIdx.y * 64  + (wave >> 2) * 32;
  const int wCol = blockIdx.x * 256 + (wave & 3) * 64;

  const unsigned char* xb = x8  + (size_t)bh * S_ * D_;
  const unsigned char* yb = yT8 + (size_t)bh * S_ * D_;

  v8i afr[2];
#pragma unroll
  for (int mt = 0; mt < 2; ++mt) {
    const int row = wRow + mt * 16 + l15;
    const u64* ap = (const u64*)(xb + (size_t)row * D_ + hf * 16);
#pragma unroll
    for (int p = 0; p < 4; ++p) {
      u64 w = ap[(p >> 1) * 4 + (p & 1)];
      afr[mt][2 * p]     = (int)(unsigned)(w & 0xffffffffull);
      afr[mt][2 * p + 1] = (int)(unsigned)(w >> 32);
    }
  }

  v8i bfr[4];
#pragma unroll
  for (int nt = 0; nt < 4; ++nt) {
    const int t = wCol + nt * 16 + l15;
    const v4i* bp = (const v4i*)(yb + (size_t)t * D_ + hf * 16);
    v4i lo = bp[0];
    v4i hi = bp[2];
    bfr[nt][0] = lo[0]; bfr[nt][1] = lo[1]; bfr[nt][2] = lo[2]; bfr[nt][3] = lo[3];
    bfr[nt][4] = hi[0]; bfr[nt][5] = hi[1]; bfr[nt][6] = hi[2]; bfr[nt][7] = hi[3];
  }

  v8i acc[2][4];
#pragma unroll
  for (int mt = 0; mt < 2; ++mt)
#pragma unroll
    for (int nt = 0; nt < 4; ++nt) {
      v8i z = {};
      acc[mt][nt] = __builtin_amdgcn_wmma_i32_16x16x64_iu8(
          true, afr[mt], true, bfr[nt], z, false, false);
    }

  const float alpha  = *alpha_p;
  const float azp    = *a_zp_p;
  const float bzp    = *b_zp_p;
  const float ozp    = *out_zp_p;
  const float oalpha = *o_alpha_p;
  const float scale  = alpha / oalpha;
  const float kzz    = (float)D_ * azp * bzp;

  float rsF[2][8];
#pragma unroll
  for (int mt = 0; mt < 2; ++mt)
#pragma unroll
    for (int v = 0; v < 8; ++v)
      rsF[mt][v] = (float)rsum[bh * S_ + wRow + mt * 16 + v + 8 * hf];

  float csF[4];
#pragma unroll
  for (int nt = 0; nt < 4; ++nt)
    csF[nt] = (float)csum[bh * S_ + wCol + nt * 16 + l15];

  float* ob = out + (size_t)bh * S_ * S_;
#pragma unroll
  for (int mt = 0; mt < 2; ++mt)
#pragma unroll
    for (int nt = 0; nt < 4; ++nt) {
      const int col = wCol + nt * 16 + l15;
#pragma unroll
      for (int v = 0; v < 8; ++v) {
        const int row = wRow + mt * 16 + v + 8 * hf;
        float q = (float)acc[mt][nt][v];
        float r = (q - bzp * rsF[mt][v] - azp * csF[nt] + kzz) * scale + ozp;
        r = fminf(fmaxf(r, -128.0f), 127.0f);
        ob[(size_t)row * S_ + col] = (float)(int)r;
      }
    }
}

extern "C" void kernel_launch(void* const* d_in, const int* in_sizes, int n_in,
                              void* d_out, int out_size, void* d_ws, size_t ws_size,
                              hipStream_t stream) {
  (void)in_sizes; (void)n_in; (void)out_size; (void)ws_size;
  const int*   x       = (const int*)d_in[0];
  const int*   y       = (const int*)d_in[1];
  const float* alpha   = (const float*)d_in[2];
  const float* a_zp    = (const float*)d_in[3];
  const float* b_zp    = (const float*)d_in[4];
  const float* out_zp  = (const float*)d_in[5];
  const float* o_alpha = (const float*)d_in[6];
  float* out = (float*)d_out;

  unsigned char* ws = (unsigned char*)d_ws;
  const size_t xyBytes = (size_t)BH_ * S_ * D_;
  unsigned char* x8  = ws;
  unsigned char* yT8 = ws + xyBytes;
  int* rsum = (int*)(ws + 2 * xyBytes);
  int* csum = rsum + (size_t)BH_ * S_;

  const int n4 = BH_ * S_ * D_ / 4;
  pack_x_kernel<<<(n4 + 255) / 256, 256, 0, stream>>>(x, x8, n4);

  const int nyT = BH_ * 4 * S_;
  pack_yT_kernel<<<(nyT + 255) / 256, 256, 0, stream>>>(y, yT8, nyT);

  const int nrows = BH_ * S_;
  rowsum_kernel<<<(nrows + 255) / 256, 256, 0, stream>>>(x, rsum, nrows);
  colsum_kernel<<<(nrows + 255) / 256, 256, 0, stream>>>(y, csum, nrows);

  dim3 grid(S_ / 256, S_ / 64, BH_);
  bmm_i8_wmma_kernel<<<grid, 256, 0, stream>>>(x8, yT8, rsum, csum, out,
                                               alpha, a_zp, b_zp, out_zp, o_alpha);
}
